// HAT3D_50972671869588
// MI455X (gfx1250) — matrix operand order repaired and checked by reading, not compiled, not run on hardware
//
#include <hip/hip_runtime.h>

typedef _Float16 v16h __attribute__((ext_vector_type(16)));
typedef _Float16 v8h  __attribute__((ext_vector_type(8)));
typedef _Float16 v4h  __attribute__((ext_vector_type(4)));
typedef float    v8f  __attribute__((ext_vector_type(8)));
typedef float    v4f  __attribute__((ext_vector_type(4)));

#define DIM 256
#define NH  8
#define HD  32
#define SCALE_QK 0.17677669529663687f   // 32^-0.5

// ---------------------------------------------------------------------------
// fp32 -> fp16 convert
// ---------------------------------------------------------------------------
__global__ void f2h_k(const float* __restrict__ in, _Float16* __restrict__ out, int n) {
    int i = blockIdx.x * 256 + threadIdx.x;
    if (i < n) out[i] = (_Float16)in[i];
}

// ---------------------------------------------------------------------------
// Positional-embedding table: 512 grid points, MLP 3 -> 512 (relu) -> 256
// ---------------------------------------------------------------------------
__global__ __launch_bounds__(256) void posemb_k(const float* __restrict__ w1,
                                                const float* __restrict__ b1,
                                                const float* __restrict__ w2,
                                                float* __restrict__ pe) {
    __shared__ float hid[512];
    int n = blockIdx.x;
    int i = n >> 6, j = (n >> 3) & 7, k = n & 7;
    float p0 = (i - 4) * 0.25f, p1 = (j - 4) * 0.25f, p2 = (k - 4) * 0.25f;
    int tid = threadIdx.x;
    for (int hh = tid; hh < 512; hh += 256) {
        float v = w1[hh * 3] * p0 + w1[hh * 3 + 1] * p1 + w1[hh * 3 + 2] * p2 + b1[hh];
        hid[hh] = v > 0.0f ? v : 0.0f;
    }
    __syncthreads();
    float acc = 0.0f;
    for (int kk = 0; kk < 512; ++kk) acc += hid[kk] * w2[tid * 512 + kk];
    pe[(size_t)n * DIM + tid] = acc;
}

// ---------------------------------------------------------------------------
// CPB table: 3375 rel-coord points, MLP 3 -> 512 (relu) -> 8
// ---------------------------------------------------------------------------
__global__ __launch_bounds__(64) void cpbtable_k(const float* __restrict__ w1,
                                                 const float* __restrict__ b1,
                                                 const float* __restrict__ w2,
                                                 float* __restrict__ table) {
    __shared__ float hid[512];
    int n = blockIdx.x;
    int i = n / 225, rem = n % 225, j = rem / 15, k = rem % 15;
    float t[3] = {(float)(i - 7), (float)(j - 7), (float)(k - 7)};
    float p[3];
    const float inv_log2_8 = 1.0f / 3.0f; // log2(8)=3
    for (int d = 0; d < 3; ++d) {
        float v = t[d] * (8.0f / 7.0f);
        float sg = (v > 0.0f) ? 1.0f : ((v < 0.0f) ? -1.0f : 0.0f);
        p[d] = sg * log2f(fabsf(v) + 1.0f) * inv_log2_8;
    }
    int tid = threadIdx.x;
    for (int hh = tid; hh < 512; hh += 64) {
        float v = w1[hh * 3] * p[0] + w1[hh * 3 + 1] * p[1] + w1[hh * 3 + 2] * p[2] + b1[hh];
        hid[hh] = v > 0.0f ? v : 0.0f;
    }
    __syncthreads();
    if (tid < 8) {
        float acc = 0.0f;
        for (int kk = 0; kk < 512; ++kk) acc += hid[kk] * w2[tid * 512 + kk];
        table[n * 8 + tid] = acc;
    }
}

// ---------------------------------------------------------------------------
// Fill PADDED bias tensor [8][Npad][Npad]:
//   pad rows/cols (>= Ntok)        -> -1e30   (softmax mask baked in)
//   global border (< ng)           -> 0
//   else                           -> 16*sigmoid(table[rel_idx])
// ---------------------------------------------------------------------------
__global__ void cpbfill_k(const float* __restrict__ table, float* __restrict__ bias,
                          int Ntok, int Npad, int ng, int total) {
    int idx = blockIdx.x * 256 + threadIdx.x;
    if (idx >= total) return;
    int nn = Npad * Npad;
    int h = idx / nn;
    int rem = idx % nn;
    int iq = rem / Npad, jk = rem % Npad;
    float v;
    if (iq >= Ntok || jk >= Ntok) {
        v = -1e30f;
    } else if (iq < ng || jk < ng) {
        v = 0.0f;
    } else {
        int p = iq - ng, q = jk - ng;
        int r0 = (p >> 6) - (q >> 6) + 7;
        int r1 = ((p >> 3) & 7) - ((q >> 3) & 7) + 7;
        int r2 = (p & 7) - (q & 7) + 7;
        int t = (r0 * 15 + r1) * 15 + r2;
        float x = table[t * 8 + h];
        v = 16.0f / (1.0f + expf(-x));
    }
    bias[idx] = v;
}

// ---------------------------------------------------------------------------
// ct de-window + hpe add : grid 512 x 256
// ---------------------------------------------------------------------------
__global__ __launch_bounds__(256) void ctdewin_k(const float* __restrict__ ct,
                                                 const float* __restrict__ pe,
                                                 float* __restrict__ out) {
    int p = blockIdx.x, c = threadIdx.x;
    int a6 = p & 1, a4 = (p >> 1) & 1, a2 = (p >> 2) & 3;
    int a5 = (p >> 4) & 1, a3 = (p >> 5) & 3, a1 = (p >> 7) & 3;
    int ip = ((((a1 * 4 + a2) * 4 + a3) * 2 + a4) * 2 + a5) * 2 + a6;
    out[p * DIM + c] = ct[ip * DIM + c] + pe[p * DIM + c];
}

// ---------------------------------------------------------------------------
// Build concat buffer xcat[64*520][256] = [ct windowed (8 tok) ; x + pe (512 tok)]
// ---------------------------------------------------------------------------
__global__ void xcat_k(const float* __restrict__ x, const float* __restrict__ pe,
                       const float* __restrict__ ctf, float* __restrict__ xcat) {
    int idx = blockIdx.x * 256 + threadIdx.x;
    if (idx >= 64 * 520 * 256) return;
    int c = idx & 255;
    int t = idx >> 8;
    int w = t / 520, r = t % 520;
    float v;
    if (r < 8) {
        int e1 = w >> 4, e3 = (w >> 2) & 3, e5 = w & 3;
        int e2 = (r >> 2) & 1, e4 = (r >> 1) & 1, e6 = r & 1;
        int p = ((((e1 * 2 + e2) * 4 + e3) * 2 + e4) * 4 + e5) * 2 + e6;
        v = ctf[p * DIM + c];
    } else {
        int n = r - 8;
        v = x[(w * 512 + n) * DIM + c] + pe[n * DIM + c];
    }
    xcat[idx] = v;
}

// ---------------------------------------------------------------------------
// LayerNorm (row = 256) -> f16 output. One wave32 per token, 8 tokens / block.
// ---------------------------------------------------------------------------
__global__ __launch_bounds__(256) void ln16_k(const float* __restrict__ x,
                                              const float* __restrict__ g,
                                              const float* __restrict__ bb,
                                              _Float16* __restrict__ out, int T) {
    int wave = threadIdx.x >> 5, lane = threadIdx.x & 31;
    int t = blockIdx.x * 8 + wave;
    if (t >= T) return;
    const float* row = x + (size_t)t * DIM;
    float v[8];
    float s = 0.0f;
    for (int i = 0; i < 8; ++i) { v[i] = row[lane * 8 + i]; s += v[i]; }
    for (int o = 16; o > 0; o >>= 1) s += __shfl_xor(s, o);
    float mean = s * (1.0f / 256.0f);
    float q = 0.0f;
    for (int i = 0; i < 8; ++i) { float d = v[i] - mean; q += d * d; }
    for (int o = 16; o > 0; o >>= 1) q += __shfl_xor(q, o);
    float rstd = rsqrtf(q * (1.0f / 256.0f) + 1e-5f);
    for (int i = 0; i < 8; ++i) {
        int c = lane * 8 + i;
        out[(size_t)t * DIM + c] = (_Float16)((v[i] - mean) * rstd * g[c] + bb[c]);
    }
}

// ---------------------------------------------------------------------------
// WMMA GEMM: C[M,N] = A[M,K] (f16) x W[N,K]^T (f16) + bias [+GELU] [+residual]
// Compile-time epilogue selection -> branch-free epilogue code.
// block = 256 thr (8 waves, 2x4), block tile 64x128, wave tile 32x32, K-step 32
// M%64==0, N%128==0, K%32==0 guaranteed by caller.
// ---------------------------------------------------------------------------
template <int GELU, int RES, int O32, int O16>
__global__ __launch_bounds__(256) void gemm16_t(const _Float16* __restrict__ A,
                                                const _Float16* __restrict__ W,
                                                const float* __restrict__ bias,
                                                const float* __restrict__ residual,
                                                float* __restrict__ outF32,
                                                _Float16* __restrict__ outF16,
                                                int M, int N, int K) {
    int tid = threadIdx.x;
    int wave = tid >> 5, lane = tid & 31;
    int lh = lane >> 4, lm = lane & 15;
    int wm = wave >> 2, wn = wave & 3;
    int bm = blockIdx.x * 64 + wm * 32;
    int bn = blockIdx.y * 128 + wn * 32;

    v8f zero = {};
    v8f acc[2][2];
    acc[0][0] = zero; acc[0][1] = zero; acc[1][0] = zero; acc[1][1] = zero;

    for (int k0 = 0; k0 < K; k0 += 32) {
        v16h a[2], b[2];
        for (int i = 0; i < 2; ++i) {
            const _Float16* pa = A + (size_t)(bm + i * 16 + lm) * K + k0;
            v8h lo = *(const v8h*)(pa + 16 * lh);
            v8h hi = *(const v8h*)(pa + 16 * lh + 8);
            for (int e = 0; e < 8; ++e) { a[i][e] = lo[e]; a[i][8 + e] = hi[e]; }
        }
        for (int j = 0; j < 2; ++j) {
            const _Float16* pw = W + (size_t)(bn + j * 16 + lm) * K + k0 + 16 * lh;
            b[j] = *(const v16h*)pw;
        }
        for (int i = 0; i < 2; ++i)
            for (int j = 0; j < 2; ++j)
                acc[i][j] = __builtin_amdgcn_wmma_f32_16x16x32_f16(
                    false, a[i], false, b[j], (short)0, acc[i][j], false, false);
    }

    for (int i = 0; i < 2; ++i)
        for (int j = 0; j < 2; ++j) {
            int col = bn + j * 16 + lm;
            float bv = bias[col];
            for (int e = 0; e < 8; ++e) {
                int row = bm + i * 16 + 8 * lh + e;
                float v = acc[i][j][e] + bv;
                if (GELU) v = 0.5f * v * (1.0f + erff(v * 0.7071067811865476f));
                size_t off = (size_t)row * N + col;
                if (RES) v += residual[off];
                if (O32) outF32[off] = v;
                if (O16) outF16[off] = (_Float16)v;
            }
        }
}

// ---------------------------------------------------------------------------
// Flash-style windowed attention, WMMA for QK^T and P*V.
// One block per (window b, head h). 128 threads = 4 waves.
// K/V staged into LDS with CDNA5 async b128 copies (ASYNCcnt path); V is then
// transposed LDS->LDS for the P*V B-operand.
// qkv:  [B*Ntok][768] f16
// bias: [8][Npad][Npad] f32 (pad entries pre-filled with -1e30 -> no masking
//       branches in the hot loop)
// out:  [B*Ntok][256] f16
// dynamic LDS = Npad*512 + 256 bytes (V-raw staging aliases the scores region).
// ---------------------------------------------------------------------------
__global__ __launch_bounds__(128) void attn_wmma_k(const _Float16* __restrict__ qkv,
                                                   const float* __restrict__ bias,
                                                   _Float16* __restrict__ out,
                                                   int Ntok, int Npad, float scale) {
    extern __shared__ char smem[];
    int b = blockIdx.x >> 3;
    int h = blockIdx.x & 7;
    int tid = threadIdx.x;
    int wave = tid >> 5, lane = tid & 31;
    int lh = lane >> 4, lm = lane & 15;

    _Float16* Ks = (_Float16*)smem;                   // [Npad][32]
    _Float16* Vt = Ks + Npad * 32;                    // [32][Npad]
    float* scoresAll = (float*)(Vt + Npad * 32);      // [4][16][Npad]
    _Float16* Pall = (_Float16*)(scoresAll + 4 * 16 * Npad); // [4][16][Npad]
    float* lsumAll = (float*)(Pall + 4 * 16 * Npad);  // [4][16]
    _Float16* Vraw = (_Float16*)scoresAll;            // [Npad][32] (aliases scores)

    // --- async staging: K rows and raw V rows are contiguous 64B runs of qkv ---
    {
        int chunks = Ntok * 4;                        // 16B chunks per row: 4
        for (int idx = tid; idx < chunks; idx += 128) {
            int n = idx >> 2, c = idx & 3;
            const _Float16* gk =
                qkv + ((size_t)(b * Ntok + n)) * 768 + 256 + h * 32 + c * 8;
            const _Float16* gv = gk + 256;
            unsigned ldsK = (unsigned)(size_t)(Ks + n * 32 + c * 8);
            unsigned ldsV = (unsigned)(size_t)(Vraw + n * 32 + c * 8);
            asm volatile("global_load_async_to_lds_b128 %0, %1, off"
                         :: "v"(ldsK), "v"(gk) : "memory");
            asm volatile("global_load_async_to_lds_b128 %0, %1, off"
                         :: "v"(ldsV), "v"(gv) : "memory");
        }
        // zero-fill pad rows (normal DS stores)
        for (int idx = tid; idx < (Npad - Ntok) * 32; idx += 128) {
            int n = Ntok + (idx >> 5), d = idx & 31;
            Ks[n * 32 + d] = (_Float16)0.0f;
            Vraw[n * 32 + d] = (_Float16)0.0f;
        }
        asm volatile("s_wait_asynccnt 0x0" ::: "memory");
        __syncthreads();
        // transpose V into Vt [32][Npad]
        for (int idx = tid; idx < Npad * 32; idx += 128) {
            int n = idx >> 5, d = idx & 31;
            Vt[d * Npad + n] = Vraw[n * 32 + d];
        }
        __syncthreads();   // Vraw (scores region) free for reuse after this
    }

    float* scoresW = scoresAll + wave * 16 * Npad;
    _Float16* Pw = Pall + wave * 16 * Npad;
    float* lsum = lsumAll + wave * 16;
    int nTiles = Npad >> 4;
    v8f zero = {};

    for (int qt = wave; qt < nTiles; qt += 4) {
        int q0 = qt * 16;
        // Q fragment (A layout), row-clamped for the pad tile
        int qr = q0 + lm; if (qr > Ntok - 1) qr = Ntok - 1;
        const _Float16* pq = qkv + ((size_t)(b * Ntok + qr)) * 768 + h * 32;
        v16h aq;
        {
            v8h lo = *(const v8h*)(pq + 16 * lh);
            v8h hi = *(const v8h*)(pq + 16 * lh + 8);
            for (int e = 0; e < 8; ++e) { aq[e] = lo[e]; aq[8 + e] = hi[e]; }
        }
        // phase 1: scores = Q K^T * scale + bias (mask pre-baked) -> LDS
        for (int kt = 0; kt < nTiles; ++kt) {
            v16h bk = *(const v16h*)(Ks + (kt * 16 + lm) * 32 + 16 * lh);
            v8f s = zero;
            s = __builtin_amdgcn_wmma_f32_16x16x32_f16(false, aq, false, bk,
                                                       (short)0, s, false, false);
            int kcol = kt * 16 + lm;
            const float* bp = bias + ((size_t)h * Npad + q0 + 8 * lh) * Npad + kcol;
            for (int e = 0; e < 8; ++e) {
                scoresW[(8 * lh + e) * Npad + kcol] = s[e] * scale + bp[(size_t)e * Npad];
            }
        }
        // phase 2: row softmax, float4-vectorized scan; P stored packed half4
        {
            int row = lane & 15;
            int st = lane >> 4;
            const float* srow = scoresW + row * Npad;
            _Float16* prow = Pw + row * Npad;
            int nch = Npad >> 2;      // float4 chunks per row
            float m = -1e30f;
            for (int c = st; c < nch; c += 2) {
                v4f xv = *(const v4f*)(srow + c * 4);
                m = fmaxf(m, fmaxf(fmaxf(xv[0], xv[1]), fmaxf(xv[2], xv[3])));
            }
            m = fmaxf(m, __shfl_xor(m, 16));
            float l = 0.0f;
            for (int c = st; c < nch; c += 2) {
                v4f xv = *(const v4f*)(srow + c * 4);
                float p0 = expf(xv[0] - m), p1 = expf(xv[1] - m);
                float p2 = expf(xv[2] - m), p3 = expf(xv[3] - m);
                l += (p0 + p1) + (p2 + p3);
                v4h pv;
                pv[0] = (_Float16)p0; pv[1] = (_Float16)p1;
                pv[2] = (_Float16)p2; pv[3] = (_Float16)p3;
                *(v4h*)(prow + c * 4) = pv;
            }
            l += __shfl_xor(l, 16);
            if (st == 0) lsum[row] = l;
        }
        // phase 3: O = P * V  (against transposed V in LDS)
        v8f o[2];
        o[0] = zero; o[1] = zero;
        for (int k0 = 0; k0 < Npad; k0 += 32) {
            v16h ap;
            {
                const _Float16* pp = Pw + (size_t)lm * Npad + k0;
                v8h lo = *(const v8h*)(pp + 16 * lh);
                v8h hi = *(const v8h*)(pp + 16 * lh + 8);
                for (int e = 0; e < 8; ++e) { ap[e] = lo[e]; ap[8 + e] = hi[e]; }
            }
            for (int j = 0; j < 2; ++j) {
                v16h bv = *(const v16h*)(Vt + (j * 16 + lm) * Npad + k0 + 16 * lh);
                o[j] = __builtin_amdgcn_wmma_f32_16x16x32_f16(false, ap, false, bv,
                                                              (short)0, o[j], false, false);
            }
        }
        // phase 4: normalize + store f16
        for (int j = 0; j < 2; ++j) {
            int d = j * 16 + lm;
            for (int e = 0; e < 8; ++e) {
                int rl = 8 * lh + e;
                int qrow = q0 + rl;
                if (qrow < Ntok) {
                    float v = o[j][e] / lsum[rl];
                    out[((size_t)(b * Ntok + qrow)) * DIM + h * 32 + d] = (_Float16)v;
                }
            }
        }
    }
}

// ---------------------------------------------------------------------------
// Final split: d_out = [x (64*512*256) | ctr (512*256)]
// ---------------------------------------------------------------------------
__global__ void outwrite_k(const float* __restrict__ xcat, float* __restrict__ out) {
    int idx = blockIdx.x * 256 + threadIdx.x;
    const int xtot = 64 * 512 * 256;
    const int total = xtot + 512 * 256;
    if (idx >= total) return;
    if (idx < xtot) {
        int c = idx & 255;
        int t = idx >> 8;
        int w = t >> 9, n = t & 511;
        out[idx] = xcat[((w * 520 + 8 + n) << 8) + c];
    } else {
        int r = idx - xtot;
        int c = r & 255;
        int t = r >> 8;
        int w = t >> 3, j = t & 7;
        out[idx] = xcat[((w * 520 + j) << 8) + c];
    }
}

// ---------------------------------------------------------------------------
// Host orchestration
// ---------------------------------------------------------------------------
extern "C" void kernel_launch(void* const* d_in, const int* in_sizes, int n_in,
                              void* d_out, int out_size, void* d_ws, size_t ws_size,
                              hipStream_t stream) {
    const float* x_in     = (const float*)d_in[0];
    const float* ct_in    = (const float*)d_in[1];
    const float* pe_w1    = (const float*)d_in[2];
    const float* pe_b1    = (const float*)d_in[3];
    const float* pe_w2    = (const float*)d_in[4];
    const float* hpe_w1   = (const float*)d_in[5];
    const float* hpe_b1   = (const float*)d_in[6];
    const float* hpe_w2   = (const float*)d_in[7];
    const float* hat_n1_g = (const float*)d_in[8];
    const float* hat_n1_b = (const float*)d_in[9];
    const float* hat_qkv_w = (const float*)d_in[10];
    const float* hat_qkv_b = (const float*)d_in[11];
    const float* hat_proj_w = (const float*)d_in[12];
    const float* hat_proj_b = (const float*)d_in[13];
    const float* hat_cpb_w1 = (const float*)d_in[14];
    const float* hat_cpb_b1 = (const float*)d_in[15];
    const float* hat_cpb_w2 = (const float*)d_in[16];
    const float* hat_n2_g = (const float*)d_in[17];
    const float* hat_n2_b = (const float*)d_in[18];
    const float* hat_fc1_w = (const float*)d_in[19];
    const float* hat_fc1_b = (const float*)d_in[20];
    const float* hat_fc2_w = (const float*)d_in[21];
    const float* hat_fc2_b = (const float*)d_in[22];
    const float* n1_g = (const float*)d_in[23];
    const float* n1_b = (const float*)d_in[24];
    const float* qkv_w = (const float*)d_in[25];
    const float* qkv_b = (const float*)d_in[26];
    const float* proj_w = (const float*)d_in[27];
    const float* proj_b = (const float*)d_in[28];
    const float* cpb_w1 = (const float*)d_in[29];
    const float* cpb_b1 = (const float*)d_in[30];
    const float* cpb_w2 = (const float*)d_in[31];
    const float* n2_g = (const float*)d_in[32];
    const float* n2_b = (const float*)d_in[33];
    const float* fc1_w = (const float*)d_in[34];
    const float* fc1_b = (const float*)d_in[35];
    const float* fc2_w = (const float*)d_in[36];
    const float* fc2_b = (const float*)d_in[37];

    const int Tx = 64 * 520;   // 33280 tokens (x path, with carrier tokens)
    const int Tc = 512;        // ct tokens

    // workspace carve (bump allocator, 256B aligned); total ~210 MB
    char* wsp = (char*)d_ws;
    auto carve = [&](size_t bytes) -> char* {
        char* p = wsp;
        wsp += (bytes + 255) & ~(size_t)255;
        return p;
    };
    float* peX      = (float*)carve((size_t)512 * DIM * 4);
    float* peCT     = (float*)carve((size_t)512 * DIM * 4);
    float* ctbuf    = (float*)carve((size_t)512 * DIM * 4);
    float* hatTab   = (float*)carve((size_t)3375 * 8 * 4);
    float* winTab   = (float*)carve((size_t)3375 * 8 * 4);
    float* hatBias  = (float*)carve((size_t)8 * 512 * 512 * 4);   // Npad=512
    float* winBias  = (float*)carve((size_t)8 * 528 * 528 * 4);   // Npad=528
    float* xcat     = (float*)carve((size_t)Tx * DIM * 4);
    _Float16* ln16  = (_Float16*)carve((size_t)Tx * DIM * 2);
    _Float16* qkv16 = (_Float16*)carve((size_t)Tx * 768 * 2);
    _Float16* att16 = (_Float16*)carve((size_t)Tx * DIM * 2);
    _Float16* h16   = (_Float16*)carve((size_t)Tx * 1024 * 2);
    _Float16* qkvW16    = (_Float16*)carve((size_t)768 * 256 * 2);
    _Float16* projW16   = (_Float16*)carve((size_t)256 * 256 * 2);
    _Float16* fc1W16    = (_Float16*)carve((size_t)1024 * 256 * 2);
    _Float16* fc2W16    = (_Float16*)carve((size_t)256 * 1024 * 2);
    _Float16* hatQkvW16 = (_Float16*)carve((size_t)768 * 256 * 2);
    _Float16* hatProjW16= (_Float16*)carve((size_t)256 * 256 * 2);
    _Float16* hatFc1W16 = (_Float16*)carve((size_t)1024 * 256 * 2);
    _Float16* hatFc2W16 = (_Float16*)carve((size_t)256 * 1024 * 2);

    auto cvt = [&](const float* src, _Float16* dst, int n) {
        f2h_k<<<(n + 255) / 256, 256, 0, stream>>>(src, dst, n);
    };
    cvt(qkv_w, qkvW16, 768 * 256);
    cvt(proj_w, projW16, 256 * 256);
    cvt(fc1_w, fc1W16, 1024 * 256);
    cvt(fc2_w, fc2W16, 256 * 1024);
    cvt(hat_qkv_w, hatQkvW16, 768 * 256);
    cvt(hat_proj_w, hatProjW16, 256 * 256);
    cvt(hat_fc1_w, hatFc1W16, 1024 * 256);
    cvt(hat_fc2_w, hatFc2W16, 256 * 1024);

    // small MLP tables
    posemb_k<<<512, 256, 0, stream>>>(pe_w1, pe_b1, pe_w2, peX);
    posemb_k<<<512, 256, 0, stream>>>(hpe_w1, hpe_b1, hpe_w2, peCT);
    cpbtable_k<<<3375, 64, 0, stream>>>(hat_cpb_w1, hat_cpb_b1, hat_cpb_w2, hatTab);
    cpbtable_k<<<3375, 64, 0, stream>>>(cpb_w1, cpb_b1, cpb_w2, winTab);
    {
        int totH = 8 * 512 * 512;
        cpbfill_k<<<(totH + 255) / 256, 256, 0, stream>>>(hatTab, hatBias, 512, 512, 0, totH);
        int totW = 8 * 528 * 528;
        cpbfill_k<<<(totW + 255) / 256, 256, 0, stream>>>(winTab, winBias, 520, 528, 8, totW);
    }

    // ---- carrier-token (hat) transformer block ----
    ctdewin_k<<<512, 256, 0, stream>>>(ct_in, peCT, ctbuf);

    ln16_k<<<Tc / 8, 256, 0, stream>>>(ctbuf, hat_n1_g, hat_n1_b, ln16, Tc);
    gemm16_t<0,0,0,1><<<dim3(Tc / 64, 768 / 128), 256, 0, stream>>>(
        ln16, hatQkvW16, hat_qkv_b, nullptr, nullptr, qkv16, Tc, 768, 256);
    {
        int Npad = 512;
        size_t sh = (size_t)Npad * 512 + 256;
        attn_wmma_k<<<1 * NH, 128, sh, stream>>>(qkv16, hatBias, att16, 512, Npad, SCALE_QK);
    }
    gemm16_t<0,1,1,0><<<dim3(Tc / 64, 256 / 128), 256, 0, stream>>>(
        att16, hatProjW16, hat_proj_b, ctbuf, ctbuf, nullptr, Tc, 256, 256);
    ln16_k<<<Tc / 8, 256, 0, stream>>>(ctbuf, hat_n2_g, hat_n2_b, ln16, Tc);
    gemm16_t<1,0,0,1><<<dim3(Tc / 64, 1024 / 128), 256, 0, stream>>>(
        ln16, hatFc1W16, hat_fc1_b, nullptr, nullptr, h16, Tc, 1024, 256);
    gemm16_t<0,1,1,0><<<dim3(Tc / 64, 256 / 128), 256, 0, stream>>>(
        h16, hatFc2W16, hat_fc2_b, ctbuf, ctbuf, nullptr, Tc, 256, 1024);

    // ---- concat [windowed ct ; x + pe] ----
    {
        int tot = 64 * 520 * 256;
        xcat_k<<<(tot + 255) / 256, 256, 0, stream>>>(x_in, peX, ctbuf, xcat);
    }

    // ---- main windowed transformer block ----
    ln16_k<<<Tx / 8, 256, 0, stream>>>(xcat, n1_g, n1_b, ln16, Tx);
    gemm16_t<0,0,0,1><<<dim3(Tx / 64, 768 / 128), 256, 0, stream>>>(
        ln16, qkvW16, qkv_b, nullptr, nullptr, qkv16, Tx, 768, 256);
    {
        int Npad = 528;
        size_t sh = (size_t)Npad * 512 + 256;
        attn_wmma_k<<<64 * NH, 128, sh, stream>>>(qkv16, winBias, att16, 520, Npad, SCALE_QK);
    }
    gemm16_t<0,1,1,0><<<dim3(Tx / 64, 256 / 128), 256, 0, stream>>>(
        att16, projW16, proj_b, xcat, xcat, nullptr, Tx, 256, 256);
    ln16_k<<<Tx / 8, 256, 0, stream>>>(xcat, n2_g, n2_b, ln16, Tx);
    gemm16_t<1,0,0,1><<<dim3(Tx / 64, 1024 / 128), 256, 0, stream>>>(
        ln16, fc1W16, fc1_b, nullptr, nullptr, h16, Tx, 1024, 256);
    gemm16_t<0,1,1,0><<<dim3(Tx / 64, 256 / 128), 256, 0, stream>>>(
        h16, fc2W16, fc2_b, xcat, xcat, nullptr, Tx, 256, 1024);

    // ---- split outputs ----
    {
        int tot = 64 * 512 * 256 + 512 * 256;
        outwrite_k<<<(tot + 255) / 256, 256, 0, stream>>>(xcat, (float*)d_out);
    }
}
